// ContrastiveLoss_22368189678030
// MI455X (gfx1250) — matrix operand order repaired and checked by reading, not compiled, not run on hardware
//
#include <hip/hip_runtime.h>
#include <hip/hip_fp16.h>
#include <math.h>

#define NROWS 8192
#define DIM   768
#define TEMP_INV 10.0f          // 1 / 0.1
#define NORM_EPS 1e-8f
#define NCLS 64
#define ROWS_PER_WG 32
#define LOG2E 1.4426950408889634f

typedef __attribute__((ext_vector_type(16))) _Float16 v16h;
typedef __attribute__((ext_vector_type(8)))  _Float16 v8h;
typedef __attribute__((ext_vector_type(8)))  float    v8f;

union V16 { v16h v; v8h h[2]; };

// raw v_exp_f32 (2^x). Safe here: |x| <= ~14.5, far from denormal range.
__device__ __forceinline__ float fast_exp2(float x) {
  return __builtin_amdgcn_exp2f(x);
}

// ---------------------------------------------------------------------------
// Kernel 0: zero the fp32 accumulators (classVec | classCount | clsLossSum).
// ---------------------------------------------------------------------------
__global__ void zero_acc(float* __restrict__ p, int n) {
  const int i = blockIdx.x * 256 + threadIdx.x;
  if (i < n) p[i] = 0.0f;
}

// ---------------------------------------------------------------------------
// Kernel 1: row-normalize x -> z (f16); accumulate per-class vector sums
// S_c (fp32, from the f16-rounded values so pos-sums match the WMMA sims)
// and per-class anchor counts.  One block per row.
// ---------------------------------------------------------------------------
__launch_bounds__(256)
__global__ void prep_normalize(const float* __restrict__ x,
                               const int* __restrict__ op_ids,
                               _Float16* __restrict__ z,
                               float* __restrict__ classVec,
                               float* __restrict__ classCount) {
  const int row = blockIdx.x;
  const float* xr = x + (size_t)row * DIM;

  const float v0 = xr[threadIdx.x];
  const float v1 = xr[threadIdx.x + 256];
  const float v2 = xr[threadIdx.x + 512];
  float ss = v0 * v0 + v1 * v1 + v2 * v2;
  #pragma unroll
  for (int off = 16; off; off >>= 1) ss += __shfl_xor(ss, off, 32);
  __shared__ float wsum[8];
  if ((threadIdx.x & 31) == 0) wsum[threadIdx.x >> 5] = ss;
  __syncthreads();
  float tot = 0.f;
  #pragma unroll
  for (int i = 0; i < 8; ++i) tot += wsum[i];

  const float inv = 1.0f / fmaxf(sqrtf(tot), NORM_EPS);
  const _Float16 h0 = (_Float16)(v0 * inv);
  const _Float16 h1 = (_Float16)(v1 * inv);
  const _Float16 h2 = (_Float16)(v2 * inv);
  _Float16* zr = z + (size_t)row * DIM;
  zr[threadIdx.x]       = h0;
  zr[threadIdx.x + 256] = h1;
  zr[threadIdx.x + 512] = h2;

  const int c = op_ids[row];
  float* cv = classVec + (size_t)c * DIM;
  atomicAdd(&cv[threadIdx.x],       (float)h0);
  atomicAdd(&cv[threadIdx.x + 256], (float)h1);
  atomicAdd(&cv[threadIdx.x + 512], (float)h2);
  if (threadIdx.x == 0) atomicAdd(&classCount[c], 1.0f);
}

// ---------------------------------------------------------------------------
// Kernel 2: posSum[i] = (z_i . S_{c(i)} - z_i . z_i) / TEMP  (exact identity
// for the positive-pair sum).  One block per row.
// ---------------------------------------------------------------------------
__launch_bounds__(256)
__global__ void pos_sums(const _Float16* __restrict__ z,
                         const int* __restrict__ op_ids,
                         const float* __restrict__ classVec,
                         float* __restrict__ posSum) {
  const int row = blockIdx.x;
  const int c = op_ids[row];
  const _Float16* zr = z + (size_t)row * DIM;
  const float* cv = classVec + (size_t)c * DIM;

  const float a0 = (float)zr[threadIdx.x];
  const float a1 = (float)zr[threadIdx.x + 256];
  const float a2 = (float)zr[threadIdx.x + 512];
  float dot  = a0 * cv[threadIdx.x] + a1 * cv[threadIdx.x + 256] +
               a2 * cv[threadIdx.x + 512];
  float self = a0 * a0 + a1 * a1 + a2 * a2;
  #pragma unroll
  for (int off = 16; off; off >>= 1) {
    dot  += __shfl_xor(dot,  off, 32);
    self += __shfl_xor(self, off, 32);
  }
  __shared__ float sd[8], sf[8];
  if ((threadIdx.x & 31) == 0) {
    sd[threadIdx.x >> 5] = dot;
    sf[threadIdx.x >> 5] = self;
  }
  __syncthreads();
  if (threadIdx.x == 0) {
    float D = 0.f, S = 0.f;
    #pragma unroll
    for (int i = 0; i < 8; ++i) { D += sd[i]; S += sf[i]; }
    posSum[row] = (D - S) * TEMP_INV;
  }
}

// ---------------------------------------------------------------------------
// Kernel 3: fused z.z^T (WMMA f16->f32) + plain exp-sum (no online max:
// |sim| <= 10 so fp32 exp-sums cannot overflow).  32 anchor rows per WG,
// two 16-row A tiles share every B fragment.  Diagonal handled only in the
// single tile per row-tile that can contain it, via a *scalar* branch
// (wave index forced uniform with readfirstlane).
// ---------------------------------------------------------------------------
__launch_bounds__(256)
__global__ void supcon_main(const _Float16* __restrict__ z,
                            const int* __restrict__ op_ids,
                            const float* __restrict__ posSum,
                            const float* __restrict__ classCount,
                            float* __restrict__ clsLossSum) {
  constexpr int LDA = DIM + 8;                   // padded half stride
  __shared__ _Float16 sA[ROWS_PER_WG * LDA];     // 32 x 768 anchor tile
  __shared__ float red_s[8][ROWS_PER_WG];

  const int tid     = threadIdx.x;
  const int wave    = __builtin_amdgcn_readfirstlane(tid >> 5);  // SGPR-uniform
  const int lane    = tid & 31;
  const int hi      = lane >> 4;
  const int lo      = lane & 15;
  const int rowBase = blockIdx.x * ROWS_PER_WG;

  for (int c = tid; c < ROWS_PER_WG * (DIM / 8); c += 256) {
    const int r = c / (DIM / 8);
    const int o = c % (DIM / 8);
    const uint4 d = ((const uint4*)(z + (size_t)(rowBase + r) * DIM))[o];
    *((uint4*)&sA[r * LDA + o * 8]) = d;
  }
  __syncthreads();

  float s0[8], s1[8];
  #pragma unroll
  for (int v = 0; v < 8; ++v) { s0[v] = 0.f; s1[v] = 0.f; }

  const int kOffA   = hi * 8;                    // ISA A-frag K layout
  const float KS    = TEMP_INV * LOG2E;          // exp(10*a) = exp2(a*KS)
  const int diagIdx = lo - 8 * hi;               // v that hits the diagonal

  for (int t = 0; t < NROWS / 128; ++t) {
    const int colBase = t * 128 + wave * 16;     // scalar (wave uniform)
    const int colG    = colBase + lo;            // B-frag: lane -> column
    const _Float16* bptr = z + (size_t)colG * DIM + hi * 8;

    v8f acc0 = {}, acc1 = {};
    #pragma unroll 4
    for (int k = 0; k < DIM; k += 32) {
      V16 a0, a1, b;
      b.h[0]  = *(const v8h*)(bptr + k);
      b.h[1]  = *(const v8h*)(bptr + k + 16);
      a0.h[0] = *(const v8h*)&sA[lo * LDA + k + kOffA];
      a0.h[1] = *(const v8h*)&sA[lo * LDA + k + kOffA + 16];
      a1.h[0] = *(const v8h*)&sA[(16 + lo) * LDA + k + kOffA];
      a1.h[1] = *(const v8h*)&sA[(16 + lo) * LDA + k + kOffA + 16];
      acc0 = __builtin_amdgcn_wmma_f32_16x16x32_f16(false, a0.v, false, b.v,
                                                    (short)0, acc0, false, false);
      acc1 = __builtin_amdgcn_wmma_f32_16x16x32_f16(false, a1.v, false, b.v,
                                                    (short)0, acc1, false, false);
    }

    if (colBase == rowBase) {                    // scalar branch, rare path
      #pragma unroll
      for (int v = 0; v < 8; ++v) {
        const float e = fast_exp2(acc0[v] * KS);
        s0[v] += (v != diagIdx) ? e : 0.f;
      }
    } else {
      #pragma unroll
      for (int v = 0; v < 8; ++v) s0[v] += fast_exp2(acc0[v] * KS);
    }
    if (colBase == rowBase + 16) {               // scalar branch, rare path
      #pragma unroll
      for (int v = 0; v < 8; ++v) {
        const float e = fast_exp2(acc1[v] * KS);
        s1[v] += (v != diagIdx) ? e : 0.f;
      }
    } else {
      #pragma unroll
      for (int v = 0; v < 8; ++v) s1[v] += fast_exp2(acc1[v] * KS);
    }
  }

  // pure-add reductions: across 16 lanes of each half-wave, then across waves
  #pragma unroll
  for (int v = 0; v < 8; ++v) {
    #pragma unroll
    for (int off = 1; off < 16; off <<= 1) {
      s0[v] += __shfl_xor(s0[v], off, 32);
      s1[v] += __shfl_xor(s1[v], off, 32);
    }
  }
  if (lo == 0) {
    #pragma unroll
    for (int v = 0; v < 8; ++v) {
      red_s[wave][v + 8 * hi]      = s0[v];
      red_s[wave][16 + v + 8 * hi] = s1[v];
    }
  }
  __syncthreads();

  if (tid < ROWS_PER_WG) {
    float S = 0.f;
    #pragma unroll
    for (int w = 0; w < 8; ++w) S += red_s[w][tid];
    const int r   = rowBase + tid;
    const int cls = op_ids[r];
    const float pc = classCount[cls] - 1.0f;     // pos_cnt = |S_c| - 1
    const float loss = (pc > 0.f) ? (-posSum[r] / pc + __logf(S)) : 0.f;
    atomicAdd(&clsLossSum[cls], loss);
  }
}

// ---------------------------------------------------------------------------
// Kernel 4: per-class mean, then mean over classes -> d_out[0].
// ---------------------------------------------------------------------------
__global__ void final_reduce(const float* __restrict__ clsLossSum,
                             const float* __restrict__ classCount,
                             const int* __restrict__ nOpPtr,
                             float* __restrict__ out) {
  const int nOp = *nOpPtr;
  const int c = threadIdx.x;                     // 64 threads = 2 waves
  float v = 0.f;
  if (c < nOp) {
    const float n = classCount[c];
    v = (n > 0.f) ? clsLossSum[c] / n : 0.f;
  }
  #pragma unroll
  for (int off = 16; off; off >>= 1) v += __shfl_xor(v, off, 32);
  __shared__ float tmp[2];
  if ((threadIdx.x & 31) == 0) tmp[threadIdx.x >> 5] = v;
  __syncthreads();
  if (threadIdx.x == 0) out[0] = (tmp[0] + tmp[1]) / (float)nOp;
}

// ---------------------------------------------------------------------------
extern "C" void kernel_launch(void* const* d_in, const int* in_sizes, int n_in,
                              void* d_out, int out_size, void* d_ws, size_t ws_size,
                              hipStream_t stream) {
  const float* x      = (const float*)d_in[0];
  const int*   op_ids = (const int*)d_in[1];
  const int*   n_op   = (const int*)d_in[2];

  _Float16* z       = (_Float16*)d_ws;
  float* classVec   = (float*)((char*)d_ws + (size_t)NROWS * DIM * sizeof(_Float16));
  float* classCount = classVec + (size_t)NCLS * DIM;
  float* clsLossSum = classCount + NCLS;
  float* posSum     = clsLossSum + NCLS;

  const int nZero = NCLS * DIM + 2 * NCLS;       // classVec|classCount|clsLossSum
  zero_acc<<<(nZero + 255) / 256, 256, 0, stream>>>(classVec, nZero);
  prep_normalize<<<NROWS, 256, 0, stream>>>(x, op_ids, z, classVec, classCount);
  pos_sums<<<NROWS, 256, 0, stream>>>(z, op_ids, classVec, posSum);
  supcon_main<<<NROWS / ROWS_PER_WG, 256, 0, stream>>>(z, op_ids, posSum,
                                                       classCount, clsLossSum);
  final_reduce<<<1, 64, 0, stream>>>(clsLossSum, classCount, n_op, (float*)d_out);
}
